// NetVLAD_45354854645905
// MI455X (gfx1250) — matrix operand order repaired and checked by reading, not compiled, not run on hardware
//
#include <hip/hip_runtime.h>
#include <hip/hip_bf16.h>

typedef __attribute__((ext_vector_type(16))) __bf16 v16bf;
typedef __attribute__((ext_vector_type(8)))  float  v8f;

union BFOp { v16bf v; unsigned u[8]; };

__device__ __forceinline__ unsigned bfpack(float a, float b) {
  __bf16 x = (__bf16)a, y = (__bf16)b;
  unsigned short ux = __builtin_bit_cast(unsigned short, x);
  unsigned short uy = __builtin_bit_cast(unsigned short, y);
  return (unsigned)ux | ((unsigned)uy << 16);
}

// reduce across lanes within each 16-lane half (rows of a WMMA C/D tile)
__device__ __forceinline__ float wsum16(float v) {
  v += __shfl_xor(v, 1, 32); v += __shfl_xor(v, 2, 32);
  v += __shfl_xor(v, 4, 32); v += __shfl_xor(v, 8, 32);
  return v;
}
__device__ __forceinline__ float wmax16(float v) {
  v = fmaxf(v, __shfl_xor(v, 1, 32)); v = fmaxf(v, __shfl_xor(v, 2, 32));
  v = fmaxf(v, __shfl_xor(v, 4, 32)); v = fmaxf(v, __shfl_xor(v, 8, 32));
  return v;
}
__device__ __forceinline__ float wsum32(float v) {
  v += __shfl_xor(v, 16, 32); return wsum16(v);
}

// ---------------- zero fill (accumulators must be re-zeroed every launch) ----
__global__ void k_zero(float* p, int n) {
  int t = blockIdx.x * 256 + threadIdx.x;
  if (t < n) p[t] = 0.f;
}

// ---------------- conv_w (64x256 f32) -> bf16 pairs [64][128] ----------------
__global__ void k_convw(const float* __restrict__ conv_w, unsigned* __restrict__ convp) {
  int t = blockIdx.x * 256 + threadIdx.x;   // 8192 pairs
  if (t < 64 * 128) convp[t] = bfpack(conv_w[2 * t], conv_w[2 * t + 1]);
}

// ---------------- K2: L2-norm + logits (WMMA bf16) + softmax -----------------
// block = 128 (4 waves), one 16-point tile per wave
__global__ void __launch_bounds__(128)
k_logits(const float* __restrict__ feat, const unsigned* __restrict__ convp,
         const float* __restrict__ conv_b, float* __restrict__ softout,
         float* __restrict__ invn, int N) {
  __shared__ unsigned xp[4][16][133];            // normalized x, bf16 pairs
  const int wid = threadIdx.x >> 5, L = threadIdx.x & 31;
  const int hh = L >> 4, r = L & 15;
  const int tile = blockIdx.x * 4 + wid;
  const int p0 = tile * 16;
  if (p0 >= N) return;

  const bool rowok = (p0 + r) < N;
  const float* rowp = feat + (size_t)(rowok ? (p0 + r) : 0) * 256 + hh * 128;

  // pass 1: sum of squares for this row-half (128 floats / lane)
  float ss = 0.f;
  for (int j = 0; j < 32; ++j) {
    float4 q = ((const float4*)rowp)[j];
    ss += q.x * q.x + q.y * q.y + q.z * q.z + q.w * q.w;
  }
  ss += __shfl_xor(ss, 16, 32);                  // combine the two halves of row r
  float scale = 1.f / fmaxf(sqrtf(ss), 1e-12f);
  if (!rowok) scale = 0.f;
  if (rowok && hh == 0) invn[p0 + r] = scale;

  // pass 2: normalized bf16 pairs into per-wave LDS tile [16 rows][128 pairs]
  for (int j = 0; j < 64; ++j) {
    float2 q = ((const float2*)rowp)[j];
    xp[wid][r][hh * 64 + j] = bfpack(q.x * scale, q.y * scale);
  }

  // logits[16 pts][64 k] = x_tile (16x256) @ conv_w^T, K-loop in chunks of 32
  v8f acc[4] = {};
  for (int cb2 = 0; cb2 < 128; cb2 += 16) {      // cb2 = pair-index base (32 chans)
    BFOp a;
    const int ab = cb2 + hh * 4;
#pragma unroll
    for (int j = 0; j < 4; ++j) { a.u[j] = xp[wid][r][ab + j]; a.u[4 + j] = xp[wid][r][ab + 8 + j]; }
#pragma unroll
    for (int kt = 0; kt < 4; ++kt) {
      const unsigned* wb = convp + (kt * 16 + r) * 128 + cb2 + hh * 4;
      uint4 q0 = ((const uint4*)wb)[0], q1 = ((const uint4*)wb)[2];
      BFOp b;
      b.u[0] = q0.x; b.u[1] = q0.y; b.u[2] = q0.z; b.u[3] = q0.w;
      b.u[4] = q1.x; b.u[5] = q1.y; b.u[6] = q1.z; b.u[7] = q1.w;
      acc[kt] = __builtin_amdgcn_wmma_f32_16x16x32_bf16(false, a.v, false, b.v,
                                                        (short)0, acc[kt], false, false);
    }
  }

  // + bias, then softmax over k=64 per point row
#pragma unroll
  for (int kt = 0; kt < 4; ++kt) {
    float cb = conv_b[kt * 16 + r];
#pragma unroll
    for (int i = 0; i < 8; ++i) acc[kt][i] += cb;
  }
#pragma unroll
  for (int i = 0; i < 8; ++i) {
    float mx = fmaxf(fmaxf(acc[0][i], acc[1][i]), fmaxf(acc[2][i], acc[3][i]));
    mx = wmax16(mx);
    float s = 0.f;
#pragma unroll
    for (int kt = 0; kt < 4; ++kt) { float e = __expf(acc[kt][i] - mx); acc[kt][i] = e; s += e; }
    s = wsum16(s);
    float rs = 1.f / s;
#pragma unroll
    for (int kt = 0; kt < 4; ++kt) acc[kt][i] *= rs;
  }
  // store soft [N][64]
#pragma unroll
  for (int i = 0; i < 8; ++i) {
    int m = i + hh * 8;
    if (p0 + m < N) {
      size_t base = (size_t)(p0 + m) * 64 + r;
#pragma unroll
      for (int kt = 0; kt < 4; ++kt) softout[base + kt * 16] = acc[kt][i];
    }
  }
}

// ---------------- K_S: S[b][k] = segment_sum(soft) ---------------------------
__global__ void __launch_bounds__(256)
k_segsum(const float* __restrict__ soft, const int* __restrict__ bid,
         float* __restrict__ S, int N) {
  __shared__ float sl[512];
  int t = threadIdx.x;
  sl[t] = 0.f; sl[t + 256] = 0.f;
  __syncthreads();
  int k = t & 63, rr = t >> 6;
  int p0 = blockIdx.x * 128;
  for (int j = 0; j < 32; ++j) {
    int p = p0 + rr * 32 + j;
    if (p < N) atomicAdd(&sl[bid[p] * 64 + k], soft[(size_t)p * 64 + k]);
  }
  __syncthreads();
  atomicAdd(&S[t], sl[t]);
  atomicAdd(&S[t + 256], sl[t + 256]);
}

// ---------------- K3: A[512][256] = W^T @ x_hat  (bf16 WMMA, split-K atomics)
// grid = (point-chunks, 8 c-slices of 32); block = 256 (8 waves)
// wave w owns bk-strips [4w,4w+4) x both 16-col tiles of this c-slice
__global__ void __launch_bounds__(256)
k_aggregate(const float* __restrict__ feat, const float* __restrict__ soft,
            const float* __restrict__ invn, const int* __restrict__ bid,
            float* __restrict__ A, int N, int iters) {
  __shared__ unsigned W2[512][17];   // masked soft, bf16 point-pairs per bk row
  __shared__ unsigned Xp[32][17];    // normalized x, bf16 point-pairs per column
  __shared__ float softs[32][65];
  __shared__ int bls[32];
  const int t = threadIdx.x, wid = t >> 5, L = t & 31, hh = L >> 4, r = L & 15;
  const int c0 = blockIdx.y * 32;
  v8f acc[4][2] = {};
  const int pstart = blockIdx.x * iters * 32;

  for (int it = 0; it < iters; ++it) {
    const int p0 = pstart + it * 32;            // uniform across block
    if (p0 >= N) break;
    // stage batch ids + soft chunk + x slice
    if (t < 32) bls[t] = (p0 + t < N) ? bid[p0 + t] : -1;
    for (int j = 0; j < 8; ++j) {
      int idx = t + 256 * j;                    // 0..2047
      int pr = idx >> 6, kk = idx & 63;
      softs[pr][kk] = (p0 + pr < N) ? soft[(size_t)(p0 + pr) * 64 + kk] : 0.f;
    }
    {
      int c = t & 31, pr = t >> 5;              // pr in 0..7
#pragma unroll
      for (int h2 = 0; h2 < 2; ++h2) {
        int pair = pr + h2 * 8;
        int p = p0 + 2 * pair;
        float x0 = 0.f, x1 = 0.f;
        if (p < N)     x0 = feat[(size_t)p * 256 + c0 + c] * invn[p];
        if (p + 1 < N) x1 = feat[(size_t)(p + 1) * 256 + c0 + c] * invn[p + 1];
        Xp[c][pair] = bfpack(x0, x1);
      }
    }
    __syncthreads();
    // build W2: W[bk][p] = soft[p][k] * (batch[p]==b), bk = b*64+k
    for (int i = 0; i < 32; ++i) {
      int e = t * 32 + i;                       // 0..8191
      int bk = e >> 4, pp = e & 15;
      int b = bk >> 6, kk = bk & 63;
      int pa = 2 * pp;
      float m0 = (bls[pa] == b) ? softs[pa][kk] : 0.f;
      float m1 = (bls[pa + 1] == b) ? softs[pa + 1][kk] : 0.f;
      W2[bk][pp] = bfpack(m0, m1);
    }
    __syncthreads();
    // per-wave WMMA: 4 strips x 2 col-tiles
    BFOp bop[2];
#pragma unroll
    for (int ct = 0; ct < 2; ++ct) {
      int c = ct * 16 + r;
#pragma unroll
      for (int j = 0; j < 8; ++j) bop[ct].u[j] = Xp[c][hh * 4 + (j & 3) + (j >> 2) * 8];
    }
#pragma unroll
    for (int s = 0; s < 4; ++s) {
      int bk = (wid * 4 + s) * 16 + r;
      BFOp a;
      int ab = hh * 4;
#pragma unroll
      for (int j = 0; j < 4; ++j) { a.u[j] = W2[bk][ab + j]; a.u[4 + j] = W2[bk][ab + 8 + j]; }
#pragma unroll
      for (int ct = 0; ct < 2; ++ct)
        acc[s][ct] = __builtin_amdgcn_wmma_f32_16x16x32_bf16(false, a.v, false, bop[ct].v,
                                                             (short)0, acc[s][ct], false, false);
    }
    __syncthreads();
  }
  // flush partials
#pragma unroll
  for (int s = 0; s < 4; ++s)
#pragma unroll
    for (int ct = 0; ct < 2; ++ct)
#pragma unroll
      for (int i = 0; i < 8; ++i) {
        int M = i + hh * 8;
        int bk = (wid * 4 + s) * 16 + M;
        int c = c0 + ct * 16 + r;
        atomicAdd(&A[bk * 256 + c], acc[s][ct][i]);
      }
}

// ---------------- K4: vlad = A - S*centroid, intra-normalize over C ----------
__global__ void __launch_bounds__(256)
k_vlad(const float* __restrict__ A, const float* __restrict__ S,
       const float* __restrict__ cent, float* __restrict__ v) {
  int wid = threadIdx.x >> 5, L = threadIdx.x & 31;
  int bk = blockIdx.x * 8 + wid;                // 0..511
  int kk = bk & 63;
  float s = S[bk];
  float val[8]; float ss = 0.f;
#pragma unroll
  for (int j = 0; j < 8; ++j) {
    int c = j * 32 + L;
    float x = A[bk * 256 + c] - s * cent[kk * 256 + c];
    val[j] = x; ss += x * x;
  }
  ss = wsum32(ss);
  float sc = 1.f / fmaxf(sqrtf(ss), 1e-12f);
#pragma unroll
  for (int j = 0; j < 8; ++j) v[bk * 256 + j * 32 + L] = val[j] * sc;
}

// ---------------- K5: out_acc = v @ fc_w^T  (bf16 WMMA, split-K) -------------
// grid = (64 red-chunks, 8); block = 256 (8 waves, one 16-col tile per wave)
__global__ void __launch_bounds__(256)
k_fc(const float* __restrict__ v, const float* __restrict__ fcw,
     float* __restrict__ outacc) {
  int wid = threadIdx.x >> 5, L = threadIdx.x & 31, hh = L >> 4, r = L & 15;
  int nt = blockIdx.y * 8 + wid;                // output tile 0..63
  int kb0 = blockIdx.x * 256;                   // 8 iters x 32
  v8f acc = {};
  for (int itr = 0; itr < 8; ++itr) {
    int kb = kb0 + itr * 32;
    BFOp a, b;
    {
      const bool valid = r < 8;                 // only 8 batch rows exist
      const float* vr = v + (size_t)r * 16384 + kb + hh * 8;
      float f[8], g[8];
#pragma unroll
      for (int j = 0; j < 8; ++j) { f[j] = valid ? vr[j] : 0.f; g[j] = valid ? vr[16 + j] : 0.f; }
#pragma unroll
      for (int j = 0; j < 4; ++j) { a.u[j] = bfpack(f[2 * j], f[2 * j + 1]); a.u[4 + j] = bfpack(g[2 * j], g[2 * j + 1]); }
    }
    {
      const float* wr = fcw + (size_t)(nt * 16 + r) * 16384 + kb + hh * 8;
#pragma unroll
      for (int j = 0; j < 8; ++j) b.u[j] = bfpack(wr[2 * j + (j >> 2) * 8], wr[2 * j + 1 + (j >> 2) * 8]);
    }
    acc = __builtin_amdgcn_wmma_f32_16x16x32_bf16(false, a.v, false, b.v,
                                                  (short)0, acc, false, false);
  }
  if (hh == 0) {                                // rows 0..7 live in lanes 0..15
#pragma unroll
    for (int i = 0; i < 8; ++i) atomicAdd(&outacc[i * 1024 + nt * 16 + r], acc[i]);
  }
}

// ---------------- K6: +bias, row L2-normalize, write output ------------------
__global__ void __launch_bounds__(256)
k_finish(const float* __restrict__ outacc, const float* __restrict__ fcb,
         float* __restrict__ out) {
  int wid = threadIdx.x >> 5, L = threadIdx.x & 31;   // row = wid (0..7)
  float val[32]; float ss = 0.f;
#pragma unroll
  for (int j = 0; j < 32; ++j) {
    int c = j * 32 + L;
    float x = outacc[wid * 1024 + c] + fcb[c];
    val[j] = x; ss += x * x;
  }
  ss = wsum32(ss);
  float sc = 1.f / fmaxf(sqrtf(ss), 1e-12f);
#pragma unroll
  for (int j = 0; j < 32; ++j) out[wid * 1024 + j * 32 + L] = val[j] * sc;
}

extern "C" void kernel_launch(void* const* d_in, const int* in_sizes, int n_in,
                              void* d_out, int out_size, void* d_ws, size_t ws_size,
                              hipStream_t stream) {
  const float* feat  = (const float*)d_in[0];
  const int*   bid   = (const int*)d_in[1];
  const float* cent  = (const float*)d_in[2];
  const float* convw = (const float*)d_in[3];
  const float* convb = (const float*)d_in[4];
  const float* fcw   = (const float*)d_in[5];
  const float* fcb   = (const float*)d_in[6];
  const int N = in_sizes[1];                    // batch_ids count

  float* ws = (float*)d_ws;
  size_t off = 0;
  float*    soft  = ws;            off += (size_t)N * 64;
  float*    invn  = ws + off;      off += (size_t)N;
  off = (off + 3) & ~(size_t)3;
  unsigned* convp = (unsigned*)(ws + off); off += 8192;   // 64*128 bf16-pairs
  float*    A     = ws + off;      off += 512 * 256;
  float*    S     = ws + off;      off += 512;
  float*    outac = ws + off;      off += 8 * 1024;
  float*    v     = ws + off;      off += 512 * 256;

  const int zn = 512 * 256 + 512 + 8 * 1024;    // A,S,outac are contiguous
  k_zero<<<(zn + 255) / 256, 256, 0, stream>>>(A, zn);
  k_convw<<<32, 256, 0, stream>>>(convw, convp);

  int ntiles = (N + 15) / 16;
  k_logits<<<(ntiles + 3) / 4, 128, 0, stream>>>(feat, convp, convb, soft, invn, N);
  k_segsum<<<(N + 127) / 128, 256, 0, stream>>>(soft, bid, S, N);

  int totIt = (N + 31) / 32;
  int G = 125;
  int iters = (totIt + G - 1) / G;
  k_aggregate<<<dim3(G, 8), 256, 0, stream>>>(feat, soft, invn, bid, A, N, iters);

  k_vlad<<<64, 256, 0, stream>>>(A, S, cent, v);
  k_fc<<<dim3(64, 8), 256, 0, stream>>>(v, fcw, outac);
  k_finish<<<1, 256, 0, stream>>>(outac, fcb, (float*)d_out);
}
